// Attention_88725434401373
// MI455X (gfx1250) — matrix operand order repaired and checked by reading, not compiled, not run on hardware
//
#include <hip/hip_runtime.h>
#include <hip/hip_bf16.h>
#include <math.h>

#define BB 2
#define SS 2048
#define DD 4096
#define HH 32
#define HDD 128

typedef __attribute__((ext_vector_type(16))) __bf16 bf16x16;
typedef __attribute__((ext_vector_type(8)))  __bf16 bf16x8;
typedef __attribute__((ext_vector_type(4)))  __bf16 bf16x4;
typedef __attribute__((ext_vector_type(8)))  float  f32x8;

// Assemble a 16-element bf16 fragment from two 16B chunks (two b128 loads).
__device__ __forceinline__ bf16x16 frag_ld(const __bf16* p0, const __bf16* p1) {
  bf16x8 a = *(const bf16x8*)p0;
  bf16x8 b = *(const bf16x8*)p1;
  bf16x16 r;
#pragma unroll
  for (int i = 0; i < 8; ++i) { r[i] = a[i]; r[i + 8] = b[i]; }
  return r;
}

// CDNA5 async global->LDS copy (no VGPR staging, tracked by ASYNCcnt).
// LDS destination address = low 32 bits of the generic pointer (AS3 offset).
__device__ __forceinline__ void async_ld16(void* lds, const void* g) {
  unsigned l = (unsigned)(unsigned long long)lds;
  asm volatile("global_load_async_to_lds_b128 %0, %1, off"
               :: "v"(l), "v"(g) : "memory");
}
__device__ __forceinline__ void wait_async_le8() {
  asm volatile("s_wait_asynccnt 0x8" ::: "memory");
}
__device__ __forceinline__ void wait_async_0() {
  asm volatile("s_wait_asynccnt 0x0" ::: "memory");
}

// ---------------------------------------------------------------------------
// Elementwise f32 -> bf16 cast (x)
// ---------------------------------------------------------------------------
__global__ __launch_bounds__(256) void cast_f32_bf16(
    const float4* __restrict__ in, __bf16* __restrict__ out, size_t n4) {
  size_t i = (size_t)blockIdx.x * 256 + threadIdx.x;
  if (i >= n4) return;
  float4 v = in[i];
  bf16x4 o;
  o[0] = (__bf16)v.x; o[1] = (__bf16)v.y; o[2] = (__bf16)v.z; o[3] = (__bf16)v.w;
  *(bf16x4*)(out + 4 * i) = o;
}

// ---------------------------------------------------------------------------
// Weight transpose + cast: w[k][n] f32 -> wt[n][k] bf16 (LDS 32x32 tile)
// ---------------------------------------------------------------------------
__global__ __launch_bounds__(256) void wtrans(
    const float* __restrict__ w, __bf16* __restrict__ wt, int Kd, int Nd) {
  __shared__ float tile[32][33];
  int n0 = blockIdx.x * 32, k0 = blockIdx.y * 32;
  int tx = threadIdx.x & 31, ty = threadIdx.x >> 5;
#pragma unroll
  for (int i = 0; i < 4; ++i) {
    int r = ty + 8 * i;
    tile[r][tx] = w[(size_t)(k0 + r) * Nd + n0 + tx];
  }
  __syncthreads();
#pragma unroll
  for (int i = 0; i < 4; ++i) {
    int r = ty + 8 * i;
    wt[(size_t)(n0 + r) * Kd + k0 + tx] = (__bf16)tile[tx][r];
  }
}

// ---------------------------------------------------------------------------
// V transpose + cast: v f32 [B,S,H,HD] -> vt bf16 [B,H,HD,S]
// ---------------------------------------------------------------------------
__global__ __launch_bounds__(256) void vtrans(
    const float* __restrict__ v, __bf16* __restrict__ vt) {
  __shared__ float tile[32][33];
  int s0 = blockIdx.x * 32, d0 = blockIdx.y * 32;
  int b = blockIdx.z >> 5, h = blockIdx.z & 31;
  int tx = threadIdx.x & 31, ty = threadIdx.x >> 5;
#pragma unroll
  for (int i = 0; i < 4; ++i) {
    int r = ty + 8 * i;
    tile[r][tx] = v[(((size_t)b * SS + s0 + r) * HH + h) * HDD + d0 + tx];
  }
  __syncthreads();
#pragma unroll
  for (int i = 0; i < 4; ++i) {
    int r = ty + 8 * i;
    vt[(((size_t)b * HH + h) * HDD + d0 + r) * SS + s0 + tx] = (__bf16)tile[tx][r];
  }
}

// ---------------------------------------------------------------------------
// bf16 GEMM:  C[M,N] = A[M,K] @ Bt[N,K]^T
// 128x128 block tile, K-step 64, 8 waves (2Mx4N), 4x2 WMMA tiles per wave.
// Global->LDS staging uses CDNA5 async copies with double buffering plus an
// L2 prefetch two tiles ahead.
// MODE 0: f32 output.  MODE 1: fused RoPE + bf16 output (for Q/K projections;
// the RoPE pair (2j,2j+1) sits in adjacent lanes of the WMMA C layout, so the
// partner element comes from one shfl_xor(.,1)).
// ---------------------------------------------------------------------------
template<int MODE>
__global__ __launch_bounds__(256) void gemm_bf16(
    const __bf16* __restrict__ A, const __bf16* __restrict__ Bt,
    void* __restrict__ Cout, const float* __restrict__ cs,
    const float* __restrict__ sn, int Nd, int Kd) {
  __shared__ __bf16 As[2][128][72];   // +8 pad
  __shared__ __bf16 Bs[2][128][72];
  const int tid = threadIdx.x;
  const int lane = tid & 31;
  const int wid = tid >> 5;
  const int wm = wid & 1, wn = wid >> 1;
  const int m0 = blockIdx.y * 128, n0 = blockIdx.x * 128;
  const int lm = lane & 15;
  const int kb0 = (lane < 16) ? 0 : 8;    // A-frag: K chunks {kb0, kb0+16}
  const int bb0 = (lane < 16) ? 0 : 8;    // B-frag: K chunks {bb0, bb0+16}

  f32x8 acc[4][2];
#pragma unroll
  for (int tm = 0; tm < 4; ++tm)
#pragma unroll
    for (int tn = 0; tn < 2; ++tn)
#pragma unroll
      for (int i = 0; i < 8; ++i) acc[tm][tn][i] = 0.0f;

  auto stage = [&](int k0, int buf) {
#pragma unroll
    for (int i = 0; i < 4; ++i) {
      int c = tid + i * 256;              // 1024 chunks of 16B per tile
      int row = c >> 3, part = (c & 7) * 8;
      async_ld16(&As[buf][row][part], A  + (size_t)(m0 + row) * Kd + k0 + part);
      async_ld16(&Bs[buf][row][part], Bt + (size_t)(n0 + row) * Kd + k0 + part);
    }
  };

  stage(0, 0);
  int buf = 0;
  for (int k0 = 0; k0 < Kd; k0 += 64) {
    if (k0 + 128 < Kd) {                  // L2 prefetch two tiles ahead
      if (tid < 128) __builtin_prefetch(A  + (size_t)(m0 + tid) * Kd + k0 + 128, 0, 1);
      else           __builtin_prefetch(Bt + (size_t)(n0 + tid - 128) * Kd + k0 + 128, 0, 1);
    }
    if (k0 + 64 < Kd) {                   // issue next tile, wait for current 8
      stage(k0 + 64, buf ^ 1);
      wait_async_le8();
    } else {
      wait_async_0();
    }
    __syncthreads();
#pragma unroll
    for (int ks = 0; ks < 2; ++ks) {
      bf16x16 bfr[2];
#pragma unroll
      for (int tn = 0; tn < 2; ++tn) {
        const __bf16* p = &Bs[buf][wn * 32 + tn * 16 + lm][ks * 32 + bb0];
        bfr[tn] = frag_ld(p, p + 16);
      }
#pragma unroll
      for (int tm = 0; tm < 4; ++tm) {
        const __bf16* p = &As[buf][wm * 64 + tm * 16 + lm][ks * 32 + kb0];
        bf16x16 afr = frag_ld(p, p + 16);
#pragma unroll
        for (int tn = 0; tn < 2; ++tn)
          acc[tm][tn] = __builtin_amdgcn_wmma_f32_16x16x32_bf16(
              false, afr, false, bfr[tn], (short)0, acc[tm][tn], false, false);
      }
    }
    __syncthreads();                      // guards refill of buf^1
    buf ^= 1;
  }

  const int half = lane >> 4;             // C layout: M = r + 8*half, N = lm
  if constexpr (MODE == 0) {
#pragma unroll
    for (int tm = 0; tm < 4; ++tm)
#pragma unroll
      for (int tn = 0; tn < 2; ++tn)
#pragma unroll
        for (int r = 0; r < 8; ++r) {
          size_t row = (size_t)m0 + wm * 64 + tm * 16 + r + 8 * half;
          size_t col = (size_t)n0 + wn * 32 + tn * 16 + lm;
          ((float*)Cout)[row * Nd + col] = acc[tm][tn][r];
        }
  } else {
    const int par = lm & 1;               // column parity within RoPE pair
#pragma unroll
    for (int tm = 0; tm < 4; ++tm)
#pragma unroll
      for (int tn = 0; tn < 2; ++tn)
#pragma unroll
        for (int r = 0; r < 8; ++r) {
          int row = m0 + wm * 64 + tm * 16 + r + 8 * half;
          int col = n0 + wn * 32 + tn * 16 + lm;
          int s = row & (SS - 1);
          int j = (col & (HDD - 1)) >> 1;
          float mine  = acc[tm][tn][r];
          float other = __shfl_xor(mine, 1);
          float c  = cs[(size_t)s * (HDD / 2) + j];
          float si = sn[(size_t)s * (HDD / 2) + j];
          float ev = par ? other : mine;  // element 2j
          float od = par ? mine : other;  // element 2j+1
          float res = par ? (ev * si + od * c) : (ev * c - od * si);
          ((__bf16*)Cout)[(size_t)row * Nd + col] = (__bf16)res;
        }
  }
}

// ---------------------------------------------------------------------------
// Flash attention (causal), bf16 WMMA, f32 accum / online softmax.
// Grid (S/128, H, B); 256 thr; wave w owns 16 q-rows. K/V chunks are
// double-buffered in LDS via async copies.
// ---------------------------------------------------------------------------
__global__ __launch_bounds__(256) void flash_attn(
    const __bf16* __restrict__ Q, const __bf16* __restrict__ Kb,
    const __bf16* __restrict__ Vt, __bf16* __restrict__ Ob) {
  __shared__ __bf16 Ks[2][128][136];    // K chunk: [key][hd], +8 pad
  __shared__ __bf16 Vs[2][128][136];    // V chunk transposed: [hd][key]
  __shared__ __bf16 Ps[8][16][136];     // per-wave P staging [qrow][key]
  const int tid = threadIdx.x, lane = tid & 31, wid = tid >> 5;
  const int q0 = blockIdx.x * 128, h = blockIdx.y, b = blockIdx.z;
  const int lm = lane & 15, half = lane >> 4;
  const int kb0 = half ? 8 : 0;
  const int bb0 = half ? 8 : 0;

  const __bf16* kbase = Kb + (((size_t)b * SS) * HH + h) * HDD;
  const __bf16* vbase = Vt + (((size_t)b * HH + h) * HDD) * SS;

  auto stageKV = [&](int kc, int buf) {
#pragma unroll
    for (int i = 0; i < 4; ++i) {
      int c = tid + i * 256;
      int row = c >> 3, part = (c & 7) * 8;
      async_ld16(&Ks[buf][row][part], kbase + (size_t)(kc + row) * DD + part);
      async_ld16(&Vs[buf][row][part], vbase + (size_t)row * SS + kc + part);
    }
  };
  stageKV(0, 0);

  // Q fragments (A layout) straight from global: row = q0 + wid*16 + lm
  const __bf16* qrow = Q + (((size_t)b * SS + q0 + wid * 16 + lm) * HH + h) * HDD;
  bf16x16 qf[4];
#pragma unroll
  for (int ks = 0; ks < 4; ++ks)
    qf[ks] = frag_ld(qrow + ks * 32 + kb0, qrow + ks * 32 + kb0 + 16);

  f32x8 o[8];
  float mrow[8], lrow[8];
#pragma unroll
  for (int r = 0; r < 8; ++r) {
    mrow[r] = -3.0e38f; lrow[r] = 0.0f;
#pragma unroll
    for (int t = 0; t < 8; ++t) o[t][r] = 0.0f;
  }

  const float sscale = 0.08838834764831845f;   // 1/sqrt(128)
  const float l2e = 1.4426950408889634f;

  int buf = 0;
  for (int kc = 0; kc <= q0; kc += 128) {
    if (kc + 128 <= q0) {                // issue next K/V chunk
      stageKV(kc + 128, buf ^ 1);
      wait_async_le8();
    } else {
      wait_async_0();
    }
    __syncthreads();

    // S = Q @ K^T  (16 x 128 per wave)
    f32x8 sacc[8];
#pragma unroll
    for (int n = 0; n < 8; ++n)
#pragma unroll
      for (int r = 0; r < 8; ++r) sacc[n][r] = 0.0f;
#pragma unroll
    for (int ks = 0; ks < 4; ++ks)
#pragma unroll
      for (int n = 0; n < 8; ++n) {
        const __bf16* p = &Ks[buf][n * 16 + lm][ks * 32 + bb0];
        bf16x16 kf = frag_ld(p, p + 16);
        sacc[n] = __builtin_amdgcn_wmma_f32_16x16x32_bf16(
            false, qf[ks], false, kf, (short)0, sacc[n], false, false);
      }

    // scale + causal mask (only the diagonal chunk is partial)
    const bool diag = (kc == q0);
#pragma unroll
    for (int n = 0; n < 8; ++n) {
      int key = kc + n * 16 + lm;
#pragma unroll
      for (int r = 0; r < 8; ++r) {
        float sv = sacc[n][r] * sscale;
        int qr = q0 + wid * 16 + r + 8 * half;
        if (diag && key > qr) sv = -1.0e9f;
        sacc[n][r] = sv;
      }
    }

    // online softmax: per-row reductions within the 16-lane half-groups
#pragma unroll
    for (int r = 0; r < 8; ++r) {
      float v = sacc[0][r];
#pragma unroll
      for (int n = 1; n < 8; ++n) v = fmaxf(v, sacc[n][r]);
      v = fmaxf(v, __shfl_xor(v, 1));
      v = fmaxf(v, __shfl_xor(v, 2));
      v = fmaxf(v, __shfl_xor(v, 4));
      v = fmaxf(v, __shfl_xor(v, 8));
      float mn = fmaxf(mrow[r], v);
      float alpha = exp2f((mrow[r] - mn) * l2e);
      mrow[r] = mn;
      lrow[r] *= alpha;
#pragma unroll
      for (int t = 0; t < 8; ++t) o[t][r] *= alpha;
      float s = 0.0f;
#pragma unroll
      for (int n = 0; n < 8; ++n) {
        float p = exp2f((sacc[n][r] - mn) * l2e);
        sacc[n][r] = p;
        s += p;
      }
      s += __shfl_xor(s, 1);
      s += __shfl_xor(s, 2);
      s += __shfl_xor(s, 4);
      s += __shfl_xor(s, 8);
      lrow[r] += s;
    }

    // P -> LDS (re-shape C layout to A layout); per-wave region, DS in-order
#pragma unroll
    for (int n = 0; n < 8; ++n)
#pragma unroll
      for (int r = 0; r < 8; ++r)
        Ps[wid][r + 8 * half][n * 16 + lm] = (__bf16)sacc[n][r];

    // O += P @ V
#pragma unroll
    for (int ks = 0; ks < 4; ++ks) {
      const __bf16* pp = &Ps[wid][lm][ks * 32 + kb0];
      bf16x16 pf = frag_ld(pp, pp + 16);
#pragma unroll
      for (int t = 0; t < 8; ++t) {
        const __bf16* vp = &Vs[buf][t * 16 + lm][ks * 32 + bb0];
        bf16x16 vf = frag_ld(vp, vp + 16);
        o[t] = __builtin_amdgcn_wmma_f32_16x16x32_bf16(
            false, pf, false, vf, (short)0, o[t], false, false);
      }
    }
    __syncthreads();                      // guards refill of buf^1
    buf ^= 1;
  }

  // normalize + write bf16 [B,S,H,HD]
  __bf16* obase = Ob + (((size_t)b * SS + q0 + wid * 16) * HH + h) * HDD;
#pragma unroll
  for (int r = 0; r < 8; ++r) {
    float inv = 1.0f / lrow[r];
#pragma unroll
    for (int t = 0; t < 8; ++t)
      obase[(size_t)(r + 8 * half) * DD + t * 16 + lm] = (__bf16)(o[t][r] * inv);
  }
}

// ---------------------------------------------------------------------------
extern "C" void kernel_launch(void* const* d_in, const int* in_sizes, int n_in,
                              void* d_out, int out_size, void* d_ws, size_t ws_size,
                              hipStream_t stream) {
  (void)in_sizes; (void)n_in; (void)out_size; (void)ws_size;
  const float* x  = (const float*)d_in[0];
  const float* wq = (const float*)d_in[1];
  const float* wk = (const float*)d_in[2];
  const float* wv = (const float*)d_in[3];
  const float* wo = (const float*)d_in[4];
  const float* fc = (const float*)d_in[5];
  const float* fs = (const float*)d_in[6];
  float* out = (float*)d_out;

  char* ws = (char*)d_ws;
  const size_t M = (size_t)BB * SS;            // 4096
  const size_t szb = M * DD * 2;               // one bf16 [4096,4096] buffer
  size_t off = 0;
  __bf16* xb  = (__bf16*)(ws + off); off += szb;
  __bf16* wqt = (__bf16*)(ws + off); off += szb;
  __bf16* wkt = (__bf16*)(ws + off); off += szb;
  __bf16* wvt = (__bf16*)(ws + off); off += szb;
  __bf16* wot = (__bf16*)(ws + off); off += szb;
  float*  tmp = (float*)(ws + off);  off += M * DD * 4;
  __bf16* qb  = (__bf16*)(ws + off); off += szb;
  __bf16* kb  = (__bf16*)(ws + off); off += szb;
  __bf16* vtb = (__bf16*)(ws + off); off += szb;
  __bf16* ab  = (__bf16*)(ws + off); off += szb;

  dim3 blk(256);
  cast_f32_bf16<<<(unsigned)(M * DD / 4 / 256), blk, 0, stream>>>(
      (const float4*)x, xb, M * DD / 4);
  dim3 tgrid(DD / 32, DD / 32);
  wtrans<<<tgrid, blk, 0, stream>>>(wq, wqt, DD, DD);
  wtrans<<<tgrid, blk, 0, stream>>>(wk, wkt, DD, DD);
  wtrans<<<tgrid, blk, 0, stream>>>(wv, wvt, DD, DD);
  wtrans<<<tgrid, blk, 0, stream>>>(wo, wot, DD, DD);

  dim3 ggrid(DD / 128, (unsigned)(M / 128));

  // Q/K projections with fused RoPE + bf16 epilogue
  gemm_bf16<1><<<ggrid, blk, 0, stream>>>(xb, wqt, qb, fc, fs, DD, DD);
  gemm_bf16<1><<<ggrid, blk, 0, stream>>>(xb, wkt, kb, fc, fs, DD, DD);

  // V projection (f32) + transpose-cast to [B,H,HD,S]
  gemm_bf16<0><<<ggrid, blk, 0, stream>>>(xb, wvt, tmp, nullptr, nullptr, DD, DD);
  vtrans<<<dim3(SS / 32, HDD / 32, BB * HH), blk, 0, stream>>>(tmp, vtb);

  flash_attn<<<dim3(SS / 128, HH, BB), blk, 0, stream>>>(qb, kb, vtb, ab);

  gemm_bf16<0><<<ggrid, blk, 0, stream>>>(ab, wot, (void*)out, nullptr, nullptr, DD, DD);
}
